// DecoderLayer_86096914416432
// MI455X (gfx1250) — matrix operand order repaired and checked by reading, not compiled, not run on hardware
//
#include <hip/hip_runtime.h>
#include <hip/hip_bf16.h>

// ---------------------------------------------------------------------------
// MI455X (gfx1250) decoder layer: bf16 WMMA GEMMs + f32 softmax/LN.
// B=8 S=256 C=2048 H=32 d=64 ENC=64 MID=4096. wave32, 256-thread blocks.
// GEMMs templated on (N,K) so row strides fold into immediate offsets.
// ---------------------------------------------------------------------------

#define B_    8
#define S_    256
#define C_    2048
#define H_    32
#define D_    64
#define ENC_  64
#define MID_  4096
#define M_    (B_ * S_)      // 2048 decoder tokens
#define ME_   (B_ * ENC_)    // 512 encoder tokens

typedef __bf16 bf16;
typedef __attribute__((ext_vector_type(8)))  __bf16 v8bf;
typedef __attribute__((ext_vector_type(16))) __bf16 v16bf;
typedef __attribute__((ext_vector_type(8)))  float  v8f;

static __device__ __forceinline__ bf16 f2bf(float f) {
  unsigned u = __builtin_bit_cast(unsigned, f);
  unsigned r = (u + 0x7FFFu + ((u >> 16) & 1u)) >> 16;   // round-to-nearest-even
  unsigned short s = (unsigned short)r;
  return __builtin_bit_cast(__bf16, s);
}

static __device__ __forceinline__ v8f vzero8() {
  v8f z = {0.f, 0.f, 0.f, 0.f, 0.f, 0.f, 0.f, 0.f};
  return z;
}

static __device__ __forceinline__ v16bf cat8(v8bf lo, v8bf hi) {
  v16bf t;
#pragma unroll
  for (int i = 0; i < 8; ++i) { t[i] = lo[i]; t[i + 8] = hi[i]; }
  return t;
}

// CDNA5 16-bit A-fragment (16x32): lanes 0-15 row M=lane, K chunks {0..7,16..23};
// lanes 16-31 same rows, K chunks {8..15,24..31}.
static __device__ __forceinline__ v16bf load_a_frag(const bf16* __restrict__ base,
                                                    size_t row_stride, int row,
                                                    int kb, int hl) {
  const bf16* ap = base + (size_t)row * row_stride + kb + 8 * hl;
  v8bf lo = *(const v8bf*)ap;
  v8bf hi = *(const v8bf*)(ap + 16);
  return cat8(lo, hi);
}

// ---------------------------------------------------------------------------
// elementwise kernels
// ---------------------------------------------------------------------------
__global__ void k_cvt_bf16(const float* __restrict__ in, bf16* __restrict__ out, int n) {
  int i = blockIdx.x * blockDim.x + threadIdx.x;
  if (i < n) out[i] = f2bf(in[i]);
}

__global__ void k_geglu(const float* __restrict__ h, float* __restrict__ out, int n) {
  int i = blockIdx.x * blockDim.x + threadIdx.x;
  if (i < n) {
    float x = h[i];
    float g = 0.5f * x * (1.0f + erff(x * 0.70710678118654752f));  // exact gelu
    out[i] = g * x;
  }
}

__global__ void k_add(const float* __restrict__ a, const float* __restrict__ b,
                      float* __restrict__ out, int n) {
  int i = blockIdx.x * blockDim.x + threadIdx.x;
  if (i < n) out[i] = a[i] + b[i];
}

// LayerNorm over last dim N, bf16 output (one row per block, 256 threads)
__global__ void k_ln_bf16(const float* __restrict__ x, const float* __restrict__ g,
                          const float* __restrict__ b, bf16* __restrict__ out, int N) {
  __shared__ float red[256];
  const int row = blockIdx.x, tid = threadIdx.x;
  const float* xr = x + (size_t)row * N;
  float s = 0.f;
  for (int i = tid; i < N; i += 256) s += xr[i];
  red[tid] = s; __syncthreads();
  for (int st = 128; st > 0; st >>= 1) { if (tid < st) red[tid] += red[tid + st]; __syncthreads(); }
  const float mu = red[0] / (float)N;
  __syncthreads();
  float v = 0.f;
  for (int i = tid; i < N; i += 256) { float d = xr[i] - mu; v += d * d; }
  red[tid] = v; __syncthreads();
  for (int st = 128; st > 0; st >>= 1) { if (tid < st) red[tid] += red[tid + st]; __syncthreads(); }
  const float rs = rsqrtf(red[0] / (float)N + 1e-5f);
  bf16* orow = out + (size_t)row * N;
  for (int i = tid; i < N; i += 256) orow[i] = f2bf((xr[i] - mu) * rs * g[i] + b[i]);
}

// out = resid + LN(t)   (f32 output)
__global__ void k_ln_add(const float* __restrict__ t, const float* __restrict__ g,
                         const float* __restrict__ b, const float* __restrict__ resid,
                         float* __restrict__ out, int N) {
  __shared__ float red[256];
  const int row = blockIdx.x, tid = threadIdx.x;
  const float* xr = t + (size_t)row * N;
  float s = 0.f;
  for (int i = tid; i < N; i += 256) s += xr[i];
  red[tid] = s; __syncthreads();
  for (int st = 128; st > 0; st >>= 1) { if (tid < st) red[tid] += red[tid + st]; __syncthreads(); }
  const float mu = red[0] / (float)N;
  __syncthreads();
  float v = 0.f;
  for (int i = tid; i < N; i += 256) { float d = xr[i] - mu; v += d * d; }
  red[tid] = v; __syncthreads();
  for (int st = 128; st > 0; st >>= 1) { if (tid < st) red[tid] += red[tid + st]; __syncthreads(); }
  const float rs = rsqrtf(red[0] / (float)N + 1e-5f);
  const float* rr = resid + (size_t)row * N;
  float* orow = out + (size_t)row * N;
  for (int i = tid; i < N; i += 256) orow[i] = rr[i] + (xr[i] - mu) * rs * g[i] + b[i];
}

// ---------------------------------------------------------------------------
// GEMM: D[M,N] = A[M,K](bf16,row-major) x Bm[K,N](bf16,row-major), f32 accum.
// 128x128 workgroup tile, 8 waves each owning 32x64 via 2x4 WMMA frags.
// N, K, SB are compile-time so strides fold into immediate offsets.
// MODE 0: store f32   MODE 1: store bf16   MODE 2: bf16 per-head transpose
//   (row=token -> (b,s), col=channel -> (h,dd);  Kt[((b*H+h)*d+dd)*SB + s])
// ---------------------------------------------------------------------------
template <int MODE, int N, int K, int SB>
__global__ __launch_bounds__(256)
void k_gemm(const bf16* __restrict__ A, const bf16* __restrict__ Bm,
            void* __restrict__ D) {
  const int lane = threadIdx.x & 31, wave = threadIdx.x >> 5;
  const int hl = lane >> 4, ln15 = lane & 15;
  const int n0 = blockIdx.x * 128 + (wave >> 2) * 64;
  const int m0 = blockIdx.y * 128 + (wave & 3) * 32;

  v8f acc[2][4];
#pragma unroll
  for (int mt = 0; mt < 2; ++mt)
#pragma unroll
    for (int nt = 0; nt < 4; ++nt) acc[mt][nt] = vzero8();

  for (int kb = 0; kb < K; kb += 32) {
    // keep HBM/L2 pipe primed two K-steps ahead (global_prefetch_b8)
    __builtin_prefetch(Bm + (size_t)(kb + 64 + lane) * N + n0, 0, 3);
    __builtin_prefetch(A + (size_t)(m0 + ln15) * K + kb + 64, 0, 3);

    v16bf a[2];
#pragma unroll
    for (int mt = 0; mt < 2; ++mt)
      a[mt] = load_a_frag(A, (size_t)K, m0 + mt * 16 + ln15, kb, hl);
    v16bf bf[4];
#pragma unroll
    for (int nt = 0; nt < 4; ++nt)
      for (int i = 0; i < 8; ++i) {
        bf[nt][i]     = Bm[(size_t)(kb + 8 * hl + i) * N + n0 + nt * 16 + ln15];
        bf[nt][i + 8] = Bm[(size_t)(kb + 16 + 8 * hl + i) * N + n0 + nt * 16 + ln15];
      }
#pragma unroll
    for (int mt = 0; mt < 2; ++mt)
#pragma unroll
      for (int nt = 0; nt < 4; ++nt)
        acc[mt][nt] = __builtin_amdgcn_wmma_f32_16x16x32_bf16(
            false, a[mt], false, bf[nt], (short)0, acc[mt][nt], false, false);
  }

#pragma unroll
  for (int mt = 0; mt < 2; ++mt)
#pragma unroll
    for (int nt = 0; nt < 4; ++nt)
#pragma unroll
      for (int i = 0; i < 8; ++i) {
        const int row = m0 + mt * 16 + hl * 8 + i;
        const int col = n0 + nt * 16 + ln15;
        const float val = acc[mt][nt][i];
        if constexpr (MODE == 0) {
          ((float*)D)[(size_t)row * N + col] = val;
        } else if constexpr (MODE == 1) {
          ((bf16*)D)[(size_t)row * N + col] = f2bf(val);
        } else {
          const int b = row / SB, s = row % SB;          // SB power-of-2 -> shifts
          const int h = col / D_, dd = col % D_;
          ((bf16*)D)[((size_t)(b * H_ + h) * D_ + dd) * SB + s] = f2bf(val);
        }
      }
}

// ---------------------------------------------------------------------------
// Attention: one (b,h) x 64-query tile per block. nK in {256, 64}.
//  Q  : bf16 [B,S,H,d]  (row-major [B*S, C])
//  Kt : bf16 [B,H,d,nK] (transposed store from K-projection GEMM)
//  V  : bf16 [B,SV,H,d]
//  O  : bf16 [B,S,H,d]
// Scores staged in padded LDS (f32), masked softmax in f32, then P*V via WMMA.
// ---------------------------------------------------------------------------
template <bool CAUSAL>
__global__ __launch_bounds__(256)
void k_attn(const bf16* __restrict__ Qp, const bf16* __restrict__ Ktp,
            const bf16* __restrict__ Vp, bf16* __restrict__ Op,
            const unsigned char* __restrict__ maskp, const int* __restrict__ tok,
            int nK, int SV) {
  __shared__ float sc[64][257];   // stride 257 -> conflict-free row scans
  const int bh = blockIdx.x, b = bh / H_, h = bh % H_;
  const int qblk = blockIdx.y * 64;
  const int lane = threadIdx.x & 31, wave = threadIdx.x >> 5;
  const int hl = lane >> 4, ln15 = lane & 15;

  const bf16* Qb  = Qp  + ((size_t)(b * S_ + qblk)) * C_ + h * D_;
  const bf16* Ktb = Ktp + ((size_t)(b * H_ + h)) * D_ * nK;
  const bf16* Vb  = Vp  + ((size_t)b * SV) * C_ + h * D_;
  bf16*       Ob  = Op  + ((size_t)(b * S_ + qblk)) * C_ + h * D_;

  // ---- phase 1: S = Q * K^T  (tiles striped over 8 waves)
  const int nTiles = nK >> 4;
  for (int t = wave; t < 4 * nTiles; t += 8) {
    const int mt = t / nTiles, nt = t % nTiles;
    v8f acc = vzero8();
#pragma unroll
    for (int ks = 0; ks < D_; ks += 32) {
      v16bf a = load_a_frag(Qb, (size_t)C_, mt * 16 + ln15, ks, hl);
      v16bf bb;
#pragma unroll
      for (int i = 0; i < 8; ++i) {
        bb[i]     = Ktb[(size_t)(ks + 8 * hl + i) * nK + nt * 16 + ln15];
        bb[i + 8] = Ktb[(size_t)(ks + 16 + 8 * hl + i) * nK + nt * 16 + ln15];
      }
      acc = __builtin_amdgcn_wmma_f32_16x16x32_bf16(false, a, false, bb,
                                                    (short)0, acc, false, false);
    }
#pragma unroll
    for (int i = 0; i < 8; ++i)
      sc[mt * 16 + hl * 8 + i][nt * 16 + ln15] = acc[i];
  }
  __syncthreads();

  // ---- phase 2: masked softmax rows (scale 1/sqrt(d) folded here)
  if (threadIdx.x < 64) {
    const int r = threadIdx.x;
    const int qg = qblk + r;
    int lim;
    if (CAUSAL) lim = tok[b * S_ + qg];  // keys k <= token_index valid
    else        lim = nK;
    float mx = -3.0e38f;
    for (int k = 0; k < nK; ++k) {
      float s = sc[r][k] * 0.125f;       // 1/sqrt(64)
      bool valid = CAUSAL ? (k <= lim) : (maskp[b * nK + k] != 0);
      s = valid ? s : -1.0e30f;
      sc[r][k] = s;
      mx = fmaxf(mx, s);
    }
    float sum = 0.f;
    for (int k = 0; k < nK; ++k) { float e = __expf(sc[r][k] - mx); sc[r][k] = e; sum += e; }
    const float inv = 1.0f / sum;
    for (int k = 0; k < nK; ++k) sc[r][k] *= inv;
  }
  __syncthreads();

  // ---- phase 3: O = P * V  (16 output tiles over 8 waves)
  for (int t = wave; t < 16; t += 8) {
    const int mt = t >> 2, nt = t & 3;
    v8f acc = vzero8();
    for (int kb = 0; kb < nK; kb += 32) {
      const int row = mt * 16 + ln15;
      v16bf a;
#pragma unroll
      for (int i = 0; i < 8; ++i) {
        a[i]     = f2bf(sc[row][kb + 8 * hl + i]);
        a[i + 8] = f2bf(sc[row][kb + 16 + 8 * hl + i]);
      }
      v16bf bb;
#pragma unroll
      for (int i = 0; i < 8; ++i) {
        bb[i]     = Vb[(size_t)(kb + 8 * hl + i) * C_ + nt * 16 + ln15];
        bb[i + 8] = Vb[(size_t)(kb + 16 + 8 * hl + i) * C_ + nt * 16 + ln15];
      }
      acc = __builtin_amdgcn_wmma_f32_16x16x32_bf16(false, a, false, bb,
                                                    (short)0, acc, false, false);
    }
#pragma unroll
    for (int i = 0; i < 8; ++i)
      Ob[(size_t)(mt * 16 + hl * 8 + i) * C_ + nt * 16 + ln15] = f2bf(acc[i]);
  }
}

// ---------------------------------------------------------------------------
// host side
// ---------------------------------------------------------------------------
extern "C" void kernel_launch(void* const* d_in, const int* in_sizes, int n_in,
                              void* d_out, int out_size, void* d_ws, size_t ws_size,
                              hipStream_t stream) {
  (void)in_sizes; (void)n_in; (void)out_size; (void)ws_size;
  const float* dec   = (const float*)d_in[0];
  const float* enc   = (const float*)d_in[1];
  const float* astat = (const float*)d_in[2];
  const unsigned char* amask = (const unsigned char*)d_in[3];
  const int*   tok   = (const int*)d_in[4];
  const float* sa_w[4] = {(const float*)d_in[5], (const float*)d_in[6],
                          (const float*)d_in[7], (const float*)d_in[8]};
  const float* ca_w[4] = {(const float*)d_in[9], (const float*)d_in[10],
                          (const float*)d_in[11], (const float*)d_in[12]};
  const float* norm_g = (const float*)d_in[13];
  const float* norm_b = (const float*)d_in[14];
  const float* glu_g1 = (const float*)d_in[15];
  const float* glu_b1 = (const float*)d_in[16];
  const float* glu_g2 = (const float*)d_in[17];
  const float* glu_b2 = (const float*)d_in[18];
  const float* glu_w1 = (const float*)d_in[19];
  const float* glu_w2 = (const float*)d_in[20];
  float* out = (float*)d_out;

  const size_t CC = (size_t)C_ * C_;        // 4M
  const size_t CM = (size_t)C_ * MID_;      // 8M
  char* wp = (char*)d_ws;
  auto alloc = [&](size_t bytes) -> void* {
    void* p = (void*)wp;
    wp += (bytes + 255) & ~(size_t)255;
    return p;
  };

  bf16* bw[8]; for (int i = 0; i < 8; ++i) bw[i] = (bf16*)alloc(CC * 2);
  bf16* bw1   = (bf16*)alloc(CM * 2);
  bf16* bw2   = (bf16*)alloc(CM * 2);
  bf16* encb  = (bf16*)alloc((size_t)ME_ * C_ * 2);
  bf16* xln   = (bf16*)alloc((size_t)M_ * C_ * 2);
  bf16* Qb    = (bf16*)alloc((size_t)M_ * C_ * 2);
  bf16* Ktb   = (bf16*)alloc((size_t)M_ * C_ * 2);
  bf16* Vb    = (bf16*)alloc((size_t)M_ * C_ * 2);
  bf16* Ob    = (bf16*)alloc((size_t)M_ * C_ * 2);
  float* t1   = (float*)alloc((size_t)M_ * C_ * 4);
  float* x1   = (float*)alloc((size_t)M_ * C_ * 4);
  bf16* xln2  = (bf16*)alloc((size_t)M_ * C_ * 2);
  bf16* Q2    = (bf16*)alloc((size_t)M_ * C_ * 2);
  bf16* eKt   = (bf16*)alloc((size_t)B_ * H_ * D_ * ENC_ * 2);
  bf16* eV    = (bf16*)alloc((size_t)ME_ * C_ * 2);
  bf16* O2    = (bf16*)alloc((size_t)M_ * C_ * 2);
  float* t2   = (float*)alloc((size_t)M_ * C_ * 4);
  float* x2   = (float*)alloc((size_t)M_ * C_ * 4);
  bf16* z     = (bf16*)alloc((size_t)M_ * C_ * 2);
  float* h1   = (float*)alloc((size_t)M_ * MID_ * 4);
  float* gact = (float*)alloc((size_t)M_ * MID_ * 4);
  bf16* z2    = (bf16*)alloc((size_t)M_ * MID_ * 2);
  float* t3   = (float*)alloc((size_t)M_ * C_ * 4);

  auto cvt = [&](const float* src, bf16* dst, size_t n) {
    k_cvt_bf16<<<dim3((unsigned)((n + 255) / 256)), 256, 0, stream>>>(src, dst, (int)n);
  };
  // bf16-cast weights (96 MB -> resident in the 192 MB L2 for all GEMMs)
  for (int i = 0; i < 4; ++i) cvt(sa_w[i], bw[i], CC);
  for (int i = 0; i < 4; ++i) cvt(ca_w[i], bw[4 + i], CC);
  cvt(glu_w1, bw1, CM);
  cvt(glu_w2, bw2, CM);
  cvt(enc, encb, (size_t)ME_ * C_);

  // ---- self-attention block
  k_ln_bf16<<<M_, 256, 0, stream>>>(dec, norm_g, norm_b, xln, C_);
  {
    dim3 g(C_ / 128, M_ / 128);
    k_gemm<1, C_, C_, S_><<<g, 256, 0, stream>>>(xln, bw[0], Qb);   // Q  [B,S,H,d]
    k_gemm<2, C_, C_, S_><<<g, 256, 0, stream>>>(xln, bw[1], Ktb);  // Kt [B,H,d,S]
    k_gemm<1, C_, C_, S_><<<g, 256, 0, stream>>>(xln, bw[2], Vb);   // V  [B,S,H,d]
  }
  k_attn<true><<<dim3(B_ * H_, S_ / 64), 256, 0, stream>>>(
      Qb, Ktb, Vb, Ob, (const unsigned char*)nullptr, tok, S_, S_);
  {
    dim3 g(C_ / 128, M_ / 128);
    k_gemm<0, C_, C_, S_><<<g, 256, 0, stream>>>(Ob, bw[3], t1);    // out-proj (f32)
  }
  k_ln_add<<<M_, 256, 0, stream>>>(t1, norm_g, norm_b, dec, x1, C_);

  // ---- cross-attention block
  k_ln_bf16<<<M_, 256, 0, stream>>>(x1, norm_g, norm_b, xln2, C_);
  {
    dim3 g(C_ / 128, M_ / 128);
    k_gemm<1, C_, C_, S_><<<g, 256, 0, stream>>>(xln2, bw[4], Q2);
  }
  {
    dim3 ge(C_ / 128, ME_ / 128);
    k_gemm<2, C_, C_, ENC_><<<ge, 256, 0, stream>>>(encb, bw[5], eKt); // enc Kt
    k_gemm<1, C_, C_, ENC_><<<ge, 256, 0, stream>>>(encb, bw[6], eV);  // enc V
  }
  k_attn<false><<<dim3(B_ * H_, S_ / 64), 256, 0, stream>>>(
      Q2, eKt, eV, O2, amask, (const int*)nullptr, ENC_, ENC_);
  {
    dim3 g(C_ / 128, M_ / 128);
    k_gemm<0, C_, C_, S_><<<g, 256, 0, stream>>>(O2, bw[7], t2);
  }
  k_ln_add<<<M_, 256, 0, stream>>>(t2, norm_g, norm_b, x1, x2, C_);

  // ---- GLU FFN
  k_ln_bf16<<<M_, 256, 0, stream>>>(x2, glu_g1, glu_b1, z, C_);
  {
    dim3 g(MID_ / 128, M_ / 128);
    k_gemm<0, MID_, C_, S_><<<g, 256, 0, stream>>>(z, bw1, h1);     // h1 = z @ w1
  }
  {
    size_t n = (size_t)M_ * MID_;
    k_geglu<<<dim3((unsigned)((n + 255) / 256)), 256, 0, stream>>>(h1, gact, (int)n);
  }
  k_ln_bf16<<<M_, 256, 0, stream>>>(gact, glu_g2, glu_b2, z2, MID_);
  {
    dim3 g(C_ / 128, M_ / 128);
    k_gemm<0, C_, MID_, S_><<<g, 256, 0, stream>>>(z2, bw2, t3);    // @ w2
  }
  {
    size_t n = (size_t)M_ * C_;
    k_add<<<dim3((unsigned)((n + 255) / 256)), 256, 0, stream>>>(x2, t3, out, (int)n);
  }

  // ---- passthrough attention_state (second tuple element)
  hipMemcpyAsync(out + (size_t)M_ * C_, astat,
                 (size_t)2 * B_ * 256 * C_ * sizeof(float),
                 hipMemcpyDeviceToDevice, stream);
}
